// SpikingNeuralNetwork_23776938951416
// MI455X (gfx1250) — matrix operand order repaired and checked by reading, not compiled, not run on hardware
//
#include <hip/hip_runtime.h>

#define B_    4096
#define NIN   1024
#define NHID  2048
#define NOUT  512
#define STEPS 25

typedef __attribute__((ext_vector_type(2)))  float  v2f;
typedef __attribute__((ext_vector_type(4)))  float  v4f;
typedef __attribute__((ext_vector_type(8)))  float  v8f;
typedef __attribute__((ext_vector_type(4)))  __bf16 v4bf;
typedef __attribute__((ext_vector_type(8)))  __bf16 v8bf;
typedef __attribute__((ext_vector_type(16))) __bf16 v16bf;

// ---------------------------------------------------------------- utilities
__global__ void zero_f32(float* __restrict__ p, long n) {
  long i = (long)blockIdx.x * blockDim.x + threadIdx.x;
  long stride = (long)gridDim.x * blockDim.x;
  for (; i < n; i += stride) p[i] = 0.0f;
}

// Split fp32 W2 into bf16 hi/lo pair: W2 == hi + lo (to bf16 rounding of residual).
// Since the A matrix (spikes) is exactly {0,1}, spk1 @ (hi+lo).T via bf16 WMMA with
// fp32 accumulation reproduces the fp32 GEMM to ~2^-16 relative accuracy.
__global__ void w2_split(const float* __restrict__ w2, __bf16* __restrict__ hi,
                         __bf16* __restrict__ lo, int n) {
  int i = blockIdx.x * blockDim.x + threadIdx.x;
  if (i < n) {
    float w = w2[i];
    __bf16 h = (__bf16)w;
    hi[i] = h;
    lo[i] = (__bf16)(w - (float)h);
  }
}

// ---------------------------------------------------------------- GEMM1 (fp32 WMMA)
// cur1[4096][2048] = x[4096][1024] @ W1.T + b1    (computed once, full fp32)
// Block: 256 threads = 8 waves as 2(M) x 4(N); wave tile 16x64; K step 4.
__global__ __launch_bounds__(256) void snn_gemm1(const float* __restrict__ x,
                                                 const float* __restrict__ w1,
                                                 const float* __restrict__ b1,
                                                 float* __restrict__ cur1) {
  const int tid = threadIdx.x;
  const int ln  = tid & 31;
  const int lr  = ln & 15;
  const int wave = tid >> 5;
  const int wm = wave >> 2;            // 0..1
  const int wn = wave & 3;             // 0..3
  const int m0 = blockIdx.x * 32 + wm * 16;
  const int n0 = blockIdx.y * 256 + wn * 64;
  const int kh = (ln & 16) ? 2 : 0;    // f32 A/B: lanes 0-15 hold K0,K1; lanes 16-31 K2,K3

  v8f acc[4] = {};
  const float* xp = x + (size_t)(m0 + lr) * NIN + kh;

  for (int k0 = 0; k0 < NIN; k0 += 4) {
    v2f a = *(const v2f*)(xp + k0);
#pragma unroll
    for (int t = 0; t < 4; ++t) {
      int col = n0 + t * 16 + lr;
      v2f b = *(const v2f*)(w1 + (size_t)col * NIN + k0 + kh);
      acc[t] = __builtin_amdgcn_wmma_f32_16x16x4_f32(false, a, false, b,
                                                     (short)0, acc[t], false, false);
    }
  }
#pragma unroll
  for (int t = 0; t < 4; ++t) {
    int col = n0 + t * 16 + lr;
    float bias = b1[col];
#pragma unroll
    for (int i = 0; i < 8; ++i) {
      int row = m0 + i + ((ln & 16) ? 8 : 0);
      cur1[(size_t)row * NHID + col] = acc[t][i] + bias;
    }
  }
}

// ---------------------------------------------------------------- step part 1
// Layer-1 leaky integrate-and-fire: mem1 RMW (L2-resident), emit bf16 spikes.
// Pure streaming, full-chip occupancy (2M threads), vectorized 4-wide.
__global__ __launch_bounds__(256) void snn_lif1(const float* __restrict__ cur1,
                                                float* __restrict__ mem1,
                                                __bf16* __restrict__ spk1) {
  long i4 = (long)blockIdx.x * blockDim.x + threadIdx.x;
  long g  = i4 * 4;                     // exact-size grid: B_*NHID/4 threads
  v4f c = *(const v4f*)(cur1 + g);
  v4f m = *(const v4f*)(mem1 + g);
  v4bf sv;
#pragma unroll
  for (int j = 0; j < 4; ++j) {
    float mp  = m[j];
    float rst = (mp > 1.0f) ? 1.0f : 0.0f;   // reset from PREVIOUS membrane
    float mn  = 0.95f * mp + c[j] - rst;     // beta * mem + cur - reset*thr
    m[j]  = mn;
    sv[j] = (mn > 1.0f) ? (__bf16)1.0f : (__bf16)0.0f;
  }
  *(v4f*)(mem1 + g) = m;
  *(v4bf*)(spk1 + g) = sv;
}

// ---------------------------------------------------------------- step part 2
// cur2 = spk1 @ (W2hi + W2lo).T + b2, fused with layer-2 LIF + output streaming.
// Barrier-free, LDS-free bf16 WMMA: A fragments are two contiguous 16B runs of a
// spike row loaded straight from global (L0/L2 hits) -> compiler can software-
// pipeline the whole K loop. Block: 512 thr = 16 waves as 4(M) x 4(N);
// wave tile 16x32 (2 C tiles, 16 acc VGPRs). Grid: (4096/64) x (512/128) = 256.
__global__ __launch_bounds__(512) void snn_gemm2(const __bf16* __restrict__ spk1,
                                                 const __bf16* __restrict__ w2hi,
                                                 const __bf16* __restrict__ w2lo,
                                                 const float* __restrict__ b2,
                                                 float* __restrict__ mem2,
                                                 float* __restrict__ spk2_out,
                                                 float* __restrict__ mem2_out) {
  const int tid = threadIdx.x;
  const int ln  = tid & 31;
  const int lr  = ln & 15;
  const int wave = tid >> 5;           // 0..15
  const int wm = wave >> 2;            // 0..3  (M sub-tile)
  const int wn = wave & 3;             // 0..3  (N sub-tile)
  const int r0 = blockIdx.x * 64 + wm * 16;
  const int c0 = blockIdx.y * 128 + wn * 32;

  // A fragment (16-bit 16x32): lane<16 -> K0..15, lane>=16 -> K16..31 (contiguous)
  const int akoff = (ln & 16) ? 16 : 0;
  const __bf16* ap = spk1 + (size_t)(r0 + lr) * NHID + akoff;
  // B fragment (16-bit 32x16): lane<16 -> K0..15, lane>=16 -> K16..31 (contiguous)
  const int bkh = (ln & 16) ? 16 : 0;

  v8f acc[2] = {};

  for (int k0 = 0; k0 < NHID; k0 += 32) {
    v8bf a0 = *(const v8bf*)(ap + k0);
    v8bf a1 = *(const v8bf*)(ap + k0 + 8);
    v16bf A = __builtin_shufflevector(a0, a1, 0, 1, 2, 3, 4, 5, 6, 7,
                                      8, 9, 10, 11, 12, 13, 14, 15);
#pragma unroll
    for (int t = 0; t < 2; ++t) {
      int col = c0 + t * 16 + lr;
      size_t bo = (size_t)col * NHID + (size_t)(k0 + bkh);
      v16bf bh = *(const v16bf*)(w2hi + bo);
      v16bf bl = *(const v16bf*)(w2lo + bo);
      acc[t] = __builtin_amdgcn_wmma_f32_16x16x32_bf16(false, A, false, bh,
                                                       (short)0, acc[t], false, false);
      acc[t] = __builtin_amdgcn_wmma_f32_16x16x32_bf16(false, A, false, bl,
                                                       (short)0, acc[t], false, false);
    }
  }

  // epilogue: bias, mem2 integrate-and-fire, stream outputs (non-temporal)
#pragma unroll
  for (int t = 0; t < 2; ++t) {
    int col = c0 + t * 16 + lr;
    float bias = b2[col];
#pragma unroll
    for (int i = 0; i < 8; ++i) {
      int row = r0 + i + ((ln & 16) ? 8 : 0);
      size_t idx = (size_t)row * NOUT + col;
      float cur2 = acc[t][i] + bias;
      float mp   = mem2[idx];
      float rst  = (mp > 1.0f) ? 1.0f : 0.0f;
      float mn   = 0.95f * mp + cur2 - rst;
      mem2[idx] = mn;
      __builtin_nontemporal_store((mn > 1.0f) ? 1.0f : 0.0f, spk2_out + idx);
      __builtin_nontemporal_store(mn, mem2_out + idx);
    }
  }
}

// ---------------------------------------------------------------- launcher
extern "C" void kernel_launch(void* const* d_in, const int* in_sizes, int n_in,
                              void* d_out, int out_size, void* d_ws, size_t ws_size,
                              hipStream_t stream) {
  (void)in_sizes; (void)n_in; (void)out_size; (void)ws_size;
  const float* x  = (const float*)d_in[0];
  const float* W1 = (const float*)d_in[1];
  const float* b1 = (const float*)d_in[2];
  const float* W2 = (const float*)d_in[3];
  const float* b2 = (const float*)d_in[4];
  float* out = (float*)d_out;

  // workspace layout (~92 MB, fits the 192 MB L2 alongside output write-combining)
  float*  cur1 = (float*)d_ws;                         // [4096][2048] fp32  32 MB
  float*  mem1 = cur1 + (size_t)B_ * NHID;             // [4096][2048] fp32  32 MB
  float*  mem2 = mem1 + (size_t)B_ * NHID;             // [4096][512]  fp32   8 MB
  __bf16* w2hi = (__bf16*)(mem2 + (size_t)B_ * NOUT);  // [512][2048]  bf16   2 MB
  __bf16* w2lo = w2hi + (size_t)NOUT * NHID;           // [512][2048]  bf16   2 MB
  __bf16* spk1 = w2lo + (size_t)NOUT * NHID;           // [4096][2048] bf16  16 MB

  // zero membrane state (mem1 and mem2 are contiguous in ws)
  long nz = (long)B_ * NHID + (long)B_ * NOUT;
  zero_f32<<<4096, 256, 0, stream>>>(mem1, nz);

  // split W2 into bf16 hi/lo
  w2_split<<<(NOUT * NHID + 255) / 256, 256, 0, stream>>>(W2, w2hi, w2lo, NOUT * NHID);

  // one-time fp32 WMMA GEMM: cur1 = x @ W1.T + b1
  dim3 g1(B_ / 32, NHID / 256);
  snn_gemm1<<<g1, 256, 0, stream>>>(x, W1, b1, cur1);

  // 25 SNN steps: LIF layer 1 (streaming) then WMMA GEMM + LIF layer 2
  const size_t stepN = (size_t)B_ * NOUT;
  const int lifBlocks = (int)((size_t)B_ * NHID / 4 / 256);   // 8192
  dim3 g2(B_ / 64, NOUT / 128);
  for (int t = 0; t < STEPS; ++t) {
    snn_lif1<<<lifBlocks, 256, 0, stream>>>(cur1, mem1, spk1);
    snn_gemm2<<<g2, 512, 0, stream>>>(
        spk1, w2hi, w2lo, b2, mem2,
        out + (size_t)t * stepN,
        out + (size_t)STEPS * stepN + (size_t)t * stepN);
  }
}
